// EntropyRegularizedAttention_91242285237020
// MI455X (gfx1250) — matrix operand order repaired and checked by reading, not compiled, not run on hardware
//
#include <hip/hip_runtime.h>

typedef __bf16 bf16;
typedef __bf16 v16bf __attribute__((ext_vector_type(16)));
typedef float  v8f   __attribute__((ext_vector_type(8)));
typedef float        f32x4 __attribute__((ext_vector_type(4)));
typedef unsigned int u32x4 __attribute__((ext_vector_type(4)));
typedef unsigned int u32x2 __attribute__((ext_vector_type(2)));

#define HIDDEN   1024
#define HEADS    16
#define HEAD_DIM 64
#define SEQ      2048
#define BATCH    2
#define MTOT     (BATCH * SEQ)   // 4096 rows total

__device__ __forceinline__ v16bf load_frag(const bf16* p0, const bf16* p1) {
    union { v16bf v; u32x4 q[2]; } u;
    u.q[0] = *(const u32x4*)p0;
    u.q[1] = *(const u32x4*)p1;
    return u.v;
}

__device__ __forceinline__ v8f wmma_bf16(v16bf a, v16bf b, v8f c) {
    return __builtin_amdgcn_wmma_f32_16x16x32_bf16(
        false, a, false, b, (short)0, c, false, false);
}

__device__ __forceinline__ v8f zero8() {
    v8f z;
    #pragma unroll
    for (int i = 0; i < 8; ++i) z[i] = 0.f;
    return z;
}

// CDNA5 async memory->LDS copy (16B per lane), tracked by ASYNCcnt.
// VDST VGPR carries the LDS byte offset (low 32 bits of the generic pointer).
__device__ __forceinline__ void async_copy_b128(void* lds_dst, const void* gsrc) {
    unsigned l = (unsigned)(unsigned long long)lds_dst;
    asm volatile("global_load_async_to_lds_b128 %0, %1, off"
                 :: "v"(l), "v"(gsrc)
                 : "memory");
}

__device__ __forceinline__ void wait_async0() {
    asm volatile("s_wait_asynccnt 0x0" ::: "memory");
}

// ---------------------------------------------------------------------------
// Tiled GEMM with bias:  Y[M,N] = X[M,K] * W[K,N] + bias
// X is fp32 or bf16 (template), W/bias fp32, Y fp32 or bf16 (template).
// 256 threads = 8 waves; block tile 128x128; K-step 32 (bf16 WMMA depth).
// Wave grid 4x2: each wave computes 32x64 = 2x4 WMMA tiles of 16x16.
// (Staging converts fp32 -> bf16 in flight, so it must use the VGPR path.)
// ---------------------------------------------------------------------------
template <bool X_IS_BF16, bool OUT_IS_F32>
__global__ __launch_bounds__(256) void gemm_bias_kernel(
    const void* __restrict__ Xv, const float* __restrict__ W,
    const float* __restrict__ bias, void* __restrict__ Yv,
    int M, int N, int K)
{
    __shared__ bf16 As[128][40];   // [m][k], +8 pad keeps rows 16B aligned
    __shared__ bf16 Bs[128][40];   // [n][k] (transposed on store)

    const int tid  = threadIdx.x;
    const int lane = tid & 31;
    const int wid  = tid >> 5;
    const int ln   = lane & 15;
    const int lhi  = lane >> 4;
    const int mw   = wid & 3;      // 0..3  -> 32-row slice
    const int nw   = wid >> 2;     // 0..1  -> 64-col slice
    const int m0   = blockIdx.y * 128;
    const int n0   = blockIdx.x * 128;

    const float* Xf = (const float*)Xv;
    const bf16*  Xb = (const bf16*)Xv;

    v8f acc[2][4];
    #pragma unroll
    for (int mt = 0; mt < 2; ++mt)
        #pragma unroll
        for (int nt = 0; nt < 4; ++nt)
            acc[mt][nt] = zero8();

    for (int kb = 0; kb < K; kb += 32) {
        // ---- stage A (128x32) and B (32x128 -> transposed) into LDS as bf16
        #pragma unroll
        for (int i = 0; i < 4; ++i) {
            int idx = tid + 256 * i;                 // 0..1023
            {   // A tile: 128 rows x 8 chunks of 4
                int row = idx >> 3, c4 = idx & 7;
                size_t off = (size_t)(m0 + row) * K + kb + c4 * 4;
                if constexpr (!X_IS_BF16) {
                    f32x4 f = *(const f32x4*)(Xf + off);
                    As[row][c4 * 4 + 0] = (bf16)f[0];
                    As[row][c4 * 4 + 1] = (bf16)f[1];
                    As[row][c4 * 4 + 2] = (bf16)f[2];
                    As[row][c4 * 4 + 3] = (bf16)f[3];
                } else {
                    *(u32x2*)&As[row][c4 * 4] = *(const u32x2*)(Xb + off);
                }
            }
            {   // B tile: 32 rows x 32 chunks of 4, store transposed
                int r = idx >> 5, c4 = idx & 31;
                f32x4 f = *(const f32x4*)(W + (size_t)(kb + r) * N + n0 + c4 * 4);
                Bs[c4 * 4 + 0][r] = (bf16)f[0];
                Bs[c4 * 4 + 1][r] = (bf16)f[1];
                Bs[c4 * 4 + 2][r] = (bf16)f[2];
                Bs[c4 * 4 + 3][r] = (bf16)f[3];
            }
        }
        if (kb + 32 < K) {  // global_prefetch_b8 of next B panel
            __builtin_prefetch(W + (size_t)(kb + 32 + lane) * N + n0 + (wid << 4), 0, 1);
        }
        __syncthreads();

        // ---- fragments (ISA 16x32 A / 32x16 B bf16 layouts)
        v16bf af[2], bfr[4];
        #pragma unroll
        for (int mt = 0; mt < 2; ++mt) {
            int row = mw * 32 + mt * 16 + ln;
            af[mt] = load_frag(&As[row][lhi * 8], &As[row][lhi * 8 + 16]);
        }
        #pragma unroll
        for (int nt = 0; nt < 4; ++nt) {
            int col = nw * 64 + nt * 16 + ln;
            bfr[nt] = load_frag(&Bs[col][lhi * 8], &Bs[col][lhi * 8 + 16]);
        }
        #pragma unroll
        for (int mt = 0; mt < 2; ++mt)
            #pragma unroll
            for (int nt = 0; nt < 4; ++nt)
                acc[mt][nt] = wmma_bf16(af[mt], bfr[nt], acc[mt][nt]);
        __syncthreads();
    }

    // ---- epilogue: bias add + store (C layout: VGPR r -> M = r + 8*lhi, N = ln)
    float* Yf = (float*)Yv;
    bf16*  Yb = (bf16*)Yv;
    #pragma unroll
    for (int mt = 0; mt < 2; ++mt)
        #pragma unroll
        for (int nt = 0; nt < 4; ++nt) {
            int col = n0 + nw * 64 + nt * 16 + ln;
            float bval = bias[col];
            #pragma unroll
            for (int r = 0; r < 8; ++r) {
                int row = m0 + mw * 32 + mt * 16 + r + lhi * 8;
                float v = acc[mt][nt][r] + bval;
                if constexpr (OUT_IS_F32) Yf[(size_t)row * N + col] = v;
                else                      Yb[(size_t)row * N + col] = (bf16)v;
            }
        }
}

// ---------------------------------------------------------------------------
// Flash attention: one block (128 thr = 4 waves) per (b, h, 64-row q-block).
// Each wave owns 16 q-rows; O accum = 4 x v8f (16x64 f32). Online softmax.
// Q and K tiles stream memory->LDS via GLOBAL_LOAD_ASYNC_TO_LDS (ASYNCcnt);
// V goes through VGPRs because it is transposed on the LDS store.
// ---------------------------------------------------------------------------
__global__ __launch_bounds__(128) void attn_kernel(
    const bf16* __restrict__ Qm, const bf16* __restrict__ Km,
    const bf16* __restrict__ Vm, bf16* __restrict__ Ctx)
{
    __shared__ bf16 sQ [64][72];   // [q][d]
    __shared__ bf16 sK [64][72];   // [key][d]
    __shared__ bf16 sVt[64][72];   // [d][key]  (transposed)
    __shared__ bf16 sP [64][72];   // [q][key]

    const int tid  = threadIdx.x;
    const int lane = tid & 31;
    const int w    = tid >> 5;
    const int ln   = lane & 15;
    const int lhi  = lane >> 4;
    const int qb   = blockIdx.x;   // 0..31
    const int h    = blockIdx.y;   // 0..15
    const int b    = blockIdx.z;   // 0..1

    const size_t baseQ = ((size_t)b * SEQ + (size_t)qb * 64) * HIDDEN + h * HEAD_DIM;

    // ---- load Q tile (64x64 bf16) asynchronously into LDS
    #pragma unroll
    for (int i = 0; i < 4; ++i) {
        int idx = tid + 128 * i;                    // 0..511
        int r = idx >> 3, c4 = idx & 7;
        async_copy_b128(&sQ[r][c4 * 8], Qm + baseQ + (size_t)r * HIDDEN + c4 * 8);
    }
    wait_async0();
    __syncthreads();

    const int qrow = w * 16 + ln;
    v16bf qf0 = load_frag(&sQ[qrow][lhi * 8],      &sQ[qrow][lhi * 8 + 16]);  // d 0..31
    v16bf qf1 = load_frag(&sQ[qrow][32 + lhi * 8], &sQ[qrow][48 + lhi * 8]);  // d 32..63

    float m_i[8], l_i[8];
    v8f o[4];
    #pragma unroll
    for (int r = 0; r < 8; ++r) { m_i[r] = -3.0e38f; l_i[r] = 0.f; }
    #pragma unroll
    for (int nt = 0; nt < 4; ++nt) o[nt] = zero8();

    const float scale = 0.125f;    // 1/sqrt(64)

    for (int kb = 0; kb < SEQ / 64; ++kb) {
        size_t baseK = ((size_t)b * SEQ + (size_t)kb * 64) * HIDDEN + h * HEAD_DIM;

        // ---- K block: async memory->LDS (no VGPR staging)
        #pragma unroll
        for (int i = 0; i < 4; ++i) {
            int idx = tid + 128 * i;
            int r = idx >> 3, c4 = idx & 7;
            async_copy_b128(&sK[r][c4 * 8], Km + baseK + (size_t)r * HIDDEN + c4 * 8);
        }
        // ---- V block: VGPR path with transpose into sVt (overlaps async K)
        #pragma unroll
        for (int i = 0; i < 4; ++i) {
            int idx = tid + 128 * i;
            int r = idx >> 3, c4 = idx & 7;
            union { u32x4 q; bf16 e[8]; } u;
            u.q = *(const u32x4*)(Vm + baseK + (size_t)r * HIDDEN + c4 * 8);
            #pragma unroll
            for (int j = 0; j < 8; ++j) sVt[c4 * 8 + j][r] = u.e[j];
        }
        wait_async0();
        __syncthreads();

        // ---- S = Q * K^T  (4 tiles x 2 WMMAs over head_dim)
        v8f s[4];
        #pragma unroll
        for (int nt = 0; nt < 4; ++nt) {
            int col = nt * 16 + ln;
            v16bf k0 = load_frag(&sK[col][lhi * 8],      &sK[col][lhi * 8 + 16]);
            v16bf k1 = load_frag(&sK[col][32 + lhi * 8], &sK[col][48 + lhi * 8]);
            v8f z = zero8();
            z = wmma_bf16(qf0, k0, z);
            z = wmma_bf16(qf1, k1, z);
            s[nt] = z;
        }

        // ---- online softmax (row reductions across the 16-lane C groups)
        #pragma unroll
        for (int r = 0; r < 8; ++r) {
            float a0 = s[0][r] * scale, a1 = s[1][r] * scale;
            float a2 = s[2][r] * scale, a3 = s[3][r] * scale;
            float mx = fmaxf(fmaxf(a0, a1), fmaxf(a2, a3));
            mx = fmaxf(mx, __shfl_xor(mx, 1, 32));
            mx = fmaxf(mx, __shfl_xor(mx, 2, 32));
            mx = fmaxf(mx, __shfl_xor(mx, 4, 32));
            mx = fmaxf(mx, __shfl_xor(mx, 8, 32));
            float mnew = fmaxf(m_i[r], mx);
            float corr = __expf(m_i[r] - mnew);
            float p0 = __expf(a0 - mnew), p1 = __expf(a1 - mnew);
            float p2 = __expf(a2 - mnew), p3 = __expf(a3 - mnew);
            float rs = p0 + p1 + p2 + p3;
            rs += __shfl_xor(rs, 1, 32);
            rs += __shfl_xor(rs, 2, 32);
            rs += __shfl_xor(rs, 4, 32);
            rs += __shfl_xor(rs, 8, 32);
            l_i[r] = l_i[r] * corr + rs;
            m_i[r] = mnew;
            s[0][r] = p0; s[1][r] = p1; s[2][r] = p2; s[3][r] = p3;
            #pragma unroll
            for (int nt = 0; nt < 4; ++nt) o[nt][r] *= corr;
        }

        // ---- spill P (bf16) to LDS to re-fragment as WMMA A operand
        #pragma unroll
        for (int nt = 0; nt < 4; ++nt)
            #pragma unroll
            for (int r = 0; r < 8; ++r)
                sP[w * 16 + r + lhi * 8][nt * 16 + ln] = (bf16)s[nt][r];
        __syncthreads();

        // ---- O += P * V
        int prow = w * 16 + ln;
        v16bf pf0 = load_frag(&sP[prow][lhi * 8],      &sP[prow][lhi * 8 + 16]);
        v16bf pf1 = load_frag(&sP[prow][32 + lhi * 8], &sP[prow][48 + lhi * 8]);
        #pragma unroll
        for (int nt = 0; nt < 4; ++nt) {
            int d = nt * 16 + ln;
            v16bf v0 = load_frag(&sVt[d][lhi * 8],      &sVt[d][lhi * 8 + 16]);
            v16bf v1 = load_frag(&sVt[d][32 + lhi * 8], &sVt[d][48 + lhi * 8]);
            o[nt] = wmma_bf16(pf0, v0, o[nt]);
            o[nt] = wmma_bf16(pf1, v1, o[nt]);
        }
        __syncthreads();
    }

    // ---- normalize and write context (bf16, [B*S, HIDDEN] layout)
    #pragma unroll
    for (int r = 0; r < 8; ++r) {
        float inv = 1.f / l_i[r];
        size_t srow = (size_t)b * SEQ + (size_t)qb * 64 + w * 16 + r + lhi * 8;
        #pragma unroll
        for (int nt = 0; nt < 4; ++nt) {
            int col = h * HEAD_DIM + nt * 16 + ln;
            Ctx[srow * HIDDEN + col] = (bf16)(o[nt][r] * inv);
        }
    }
}

// ---------------------------------------------------------------------------
extern "C" void kernel_launch(void* const* d_in, const int* in_sizes, int n_in,
                              void* d_out, int out_size, void* d_ws, size_t ws_size,
                              hipStream_t stream)
{
    const float* x  = (const float*)d_in[0];
    const float* Wq = (const float*)d_in[1];
    const float* bq = (const float*)d_in[2];
    const float* Wk = (const float*)d_in[3];
    const float* bk = (const float*)d_in[4];
    const float* Wv = (const float*)d_in[5];
    const float* bv = (const float*)d_in[6];
    const float* Wo = (const float*)d_in[7];
    const float* bo = (const float*)d_in[8];

    const size_t BUF = (size_t)MTOT * HIDDEN;   // 4M elems
    bf16* Q = (bf16*)d_ws;
    bf16* K = Q + BUF;
    bf16* V = K + BUF;
    bf16* C = V + BUF;

    dim3 gg(HIDDEN / 128, MTOT / 128);          // (8, 32)
    gemm_bias_kernel<false, false><<<gg, 256, 0, stream>>>(x, Wq, bq, Q, MTOT, HIDDEN, HIDDEN);
    gemm_bias_kernel<false, false><<<gg, 256, 0, stream>>>(x, Wk, bk, K, MTOT, HIDDEN, HIDDEN);
    gemm_bias_kernel<false, false><<<gg, 256, 0, stream>>>(x, Wv, bv, V, MTOT, HIDDEN, HIDDEN);

    attn_kernel<<<dim3(SEQ / 64, HEADS, BATCH), 128, 0, stream>>>(Q, K, V, C);

    gemm_bias_kernel<true, true><<<gg, 256, 0, stream>>>(C, Wo, bo, d_out, MTOT, HIDDEN, HIDDEN);
}
